// CorrBlock2_75299366633998
// MI455X (gfx1250) — matrix operand order repaired and checked by reading, not compiled, not run on hardware
//
#include <hip/hip_runtime.h>
#include <hip/hip_bf16.h>
#include <math.h>

typedef __attribute__((ext_vector_type(16))) _Float16 v16h;
typedef __attribute__((ext_vector_type(8)))  float    v8f;

#define BB   2
#define NN   16384
#define KK   128
#define DKK  28
#define NBP  (BB*NN)

// A-operand k order used here (contiguous, same as w2p/owp packing): lane half h (0/1), element j -> K index
__device__ __forceinline__ int amap(int h, int j) {
  return j + 16 * h;
}

// ---------------- prep: fold w1 levels, pack f16 weight tiles ----------------
__global__ __launch_bounds__(256) void kprep(
    const float* __restrict__ w1, const float* __restrict__ w2,
    const float* __restrict__ ow, float* __restrict__ w1eff,
    _Float16* __restrict__ w2p, _Float16* __restrict__ owp) {
  int tid = threadIdx.x;
  for (int t = tid; t < 128 * 27; t += 256) {
    int c = t / 27, i = t % 27;
    w1eff[t] = w1[c * 81 + i] + w1[c * 81 + 27 + i] + w1[c * 81 + 54 + i];
  }
  // w2: [64,128] -> B tiles q=kt*4+nt, B[k,c]=w2[c][k]; B layout: K = j + 16*(lane/16)
  for (int t = tid; t < 16 * 32 * 16; t += 256) {
    int q = t >> 9, lane = (t >> 4) & 31, j = t & 15;
    int kt = q >> 2, nt = q & 3;
    int k = kt * 32 + j + 16 * (lane >> 4);
    int c = nt * 16 + (lane & 15);
    w2p[t] = (_Float16)w2[c * 128 + k];
  }
  // ow: [64,64]
  for (int t = tid; t < 8 * 32 * 16; t += 256) {
    int q = t >> 9, lane = (t >> 4) & 31, j = t & 15;
    int kt = q >> 2, nt = q & 3;
    int k = kt * 32 + j + 16 * (lane >> 4);
    int c = nt * 16 + (lane & 15);
    owp[t] = (_Float16)ow[c * 64 + k];
  }
}

// ---------------- per-point heavy kernel: one wave per point ----------------
__global__ __launch_bounds__(256) void kpoint(
    const float* __restrict__ coords, const float* __restrict__ corr,
    const float* __restrict__ xyz2, const float* __restrict__ dflow,
    const float* __restrict__ b1, const float* __restrict__ w1eff,
    float* __restrict__ x_pre, float* __restrict__ fws,
    float* __restrict__ gsum, float* __restrict__ gsq,
    float* __restrict__ S1, float* __restrict__ S2) {
  __shared__ float sdist[8][128];
  __shared__ float hadd[8][28];
  __shared__ float hcnt[8][28];
  __shared__ float scorr[8][28];
  __shared__ float sw1[128 * 27];
  __shared__ float bsum[8], bsq[8], bS[14];

  int tid = threadIdx.x;
  int wave = tid >> 5, lane = tid & 31;
  int p = blockIdx.x * 8 + wave;         // global point, b uniform per block
  int b = p >> 14;
  int n = p & (NN - 1);

  if (tid < 8)  { bsum[tid] = 0.f; bsq[tid] = 0.f; }
  if (tid < 14) bS[tid] = 0.f;
  if (lane < 28) { hadd[wave][lane] = 0.f; hcnt[wave][lane] = 0.f; }
  for (int t = tid; t < 128 * 27; t += 256) sw1[t] = w1eff[t];

  // ---- adaptive frame: unit axes (axis scale cancels in projection) ----
  size_t fo  = ((size_t)b * NN + n) * 3;
  size_t fo2 = fo + (size_t)BB * NN * 3;
  float f1x = dflow[fo],  f1y = dflow[fo + 1],  f1z = dflow[fo + 2];
  float f2x = dflow[fo2], f2y = dflow[fo2 + 1], f2z = dflow[fo2 + 2];
  float n1 = sqrtf(f1x * f1x + f1y * f1y + f1z * f1z);
  float n2 = sqrtf(f2x * f2x + f2y * f2y + f2z * f2z);
  float dot12 = f1x * f2x + f1y * f2y + f1z * f2z;
  float cos1 = dot12 / (fmaxf(n1, 1e-8f) * fmaxf(n2, 1e-8f));
  float t12 = cos1 * n2 / n1;
  float r2px = f2x - t12 * f1x, r2py = f2y - t12 * f1y, r2pz = f2z - t12 * f1z;
  float n2p = sqrtf(r2px * r2px + r2py * r2py + r2pz * r2pz);
  float cxp = f1y * f2z - f1z * f2y;
  float cyp = f1z * f2x - f1x * f2z;
  float czp = f1x * f2y - f1y * f2x;
  float n3 = sqrtf(cxp * cxp + cyp * cyp + czp * czp);
  float u1x = f1x / n1,  u1y = f1y / n1,  u1z = f1z / n1;
  float u2x = r2px / n2p, u2y = r2py / n2p, u2z = r2pz / n2p;
  float u3x = cxp / n3,  u3y = cyp / n3,  u3z = czp / n3;

  size_t co = ((size_t)b * NN + n) * 3;
  float cx = coords[co], cy = coords[co + 1], cz = coords[co + 2];

  __syncthreads();

  float dfx[4], dfy[4], dfz[4], crv[4], dstv[4];
  #pragma unroll
  for (int kk = 0; kk < 4; kk++) {
    int k = kk * 32 + lane;
    size_t base = (size_t)p * KK + k;
    float x = xyz2[base * 3 + 0] - cx;
    float y = xyz2[base * 3 + 1] - cy;
    float z = xyz2[base * 3 + 2] - cz;
    float cv = corr[base];
    float d1 = x * u1x + y * u1y + z * u1z;
    float d2 = x * u2x + y * u2y + z * u2z;
    float d3 = x * u3x + y * u3y + z * u3z;
    float r1 = rintf(d1), r2 = rintf(d2), r3 = rintf(d3);  // round-half-even like jnp.round
    float dd = x * x + y * y + z * z;
    dfx[kk] = x; dfy[kk] = y; dfz[kk] = z; crv[kk] = cv; dstv[kk] = dd;
    sdist[wave][k] = dd;
    if (fabsf(r1) <= 1.f && fabsf(r2) <= 1.f && fabsf(r3) <= 1.f) {
      int cu = ((int)r1) * 9 + ((int)r2) * 3 + ((int)r3) + 13;   // (r+1) linearized
      atomicAdd(&hadd[wave][cu], cv);
      atomicAdd(&hcnt[wave][cu], 1.f);
    }
  }
  __syncthreads();
  if (lane < 27)
    scorr[wave][lane] = hadd[wave][lane] / fmaxf(hcnt[wave][lane], 1.f);
  __syncthreads();

  // ---- x128 = w1eff @ corr27 + b1 ; accumulate GN1 group sums ----
  #pragma unroll
  for (int j = 0; j < 4; j++) {
    int c = lane + 32 * j;
    float acc = b1[c];
    #pragma unroll
    for (int i = 0; i < 27; i++) acc += sw1[c * 27 + i] * scorr[wave][i];
    x_pre[(size_t)p * 128 + c] = acc;
    float s = acc, q = acc * acc;
    #pragma unroll
    for (int m = 1; m < 16; m <<= 1) { s += __shfl_xor(s, m); q += __shfl_xor(q, m); }
    if ((lane & 15) == 0) {
      int g = 2 * j + (lane >> 4);         // 16 channels per group
      atomicAdd(&bsum[g], s); atomicAdd(&bsq[g], q);
    }
  }

  // ---- exact top-28 (lax.top_k tie semantics: lower index wins) ----
  float s10 = 0.f, s11 = 0.f, s12 = 0.f, s13 = 0.f;
  float s2p[10] = {0.f,0.f,0.f,0.f,0.f,0.f,0.f,0.f,0.f,0.f};
  #pragma unroll
  for (int kk = 0; kk < 4; kk++) {
    int k = kk * 32 + lane;
    float di = dstv[kk];
    int rank = 0;
    #pragma unroll 4
    for (int j = 0; j < 128; j++) {
      float dj = sdist[wave][j];
      rank += (dj < di || (dj == di && j < k)) ? 1 : 0;
    }
    if (rank < DKK) {
      float a0 = crv[kk], a1v = dfx[kk], a2v = dfy[kk], a3v = dfz[kk];
      *(float4*)&fws[((size_t)p * DKK + rank) * 4] = make_float4(a0, a1v, a2v, a3v);
      s10 += a0; s11 += a1v; s12 += a2v; s13 += a3v;
      s2p[0] += a0*a0;  s2p[1] += a0*a1v; s2p[2] += a0*a2v; s2p[3] += a0*a3v;
      s2p[4] += a1v*a1v; s2p[5] += a1v*a2v; s2p[6] += a1v*a3v;
      s2p[7] += a2v*a2v; s2p[8] += a2v*a3v; s2p[9] += a3v*a3v;
    }
  }
  #pragma unroll
  for (int m = 1; m < 32; m <<= 1) {
    s10 += __shfl_xor(s10, m); s11 += __shfl_xor(s11, m);
    s12 += __shfl_xor(s12, m); s13 += __shfl_xor(s13, m);
    #pragma unroll
    for (int u = 0; u < 10; u++) s2p[u] += __shfl_xor(s2p[u], m);
  }
  if (lane == 0) {
    atomicAdd(&bS[0], s10); atomicAdd(&bS[1], s11);
    atomicAdd(&bS[2], s12); atomicAdd(&bS[3], s13);
    #pragma unroll
    for (int u = 0; u < 10; u++) atomicAdd(&bS[4 + u], s2p[u]);
  }
  __syncthreads();
  if (tid < 8) { atomicAdd(&gsum[b * 8 + tid], bsum[tid]); atomicAdd(&gsq[b * 8 + tid], bsq[tid]); }
  if (tid < 4) atomicAdd(&S1[b * 4 + tid], bS[tid]);
  if (tid >= 4 && tid < 14) atomicAdd(&S2[b * 10 + (tid - 4)], bS[tid]);
}

// ---------------- finalize GN affines ----------------
__global__ __launch_bounds__(128) void kfin(
    const float* __restrict__ gsum, const float* __restrict__ gsq,
    const float* __restrict__ S1, const float* __restrict__ S2,
    const float* __restrict__ gn1g, const float* __restrict__ gn1b,
    const float* __restrict__ gn2g, const float* __restrict__ gn2b,
    const float* __restrict__ kw, const float* __restrict__ kb,
    float* __restrict__ A1, float* __restrict__ C1,
    float* __restrict__ A2, float* __restrict__ C2) {
  __shared__ float m1[8], i1[8], sy[64], syy[64], m2[8], i2[8];
  int tid = threadIdx.x;
  const float CNT1 = 16.0f * NN;                 // elems per GN1 group
  const float CNT2 = (float)NN * (float)DKK;     // (n,k) elems per channel (GN2)
  for (int b = 0; b < BB; b++) {
    if (tid < 8) {
      float mean = gsum[b * 8 + tid] / CNT1;
      float var  = gsq[b * 8 + tid] / CNT1 - mean * mean;
      m1[tid] = mean; i1[tid] = rsqrtf(var + 1e-5f);
    }
    if (tid < 64) {
      int c = tid;
      float k0 = kw[c*4], k1 = kw[c*4+1], k2 = kw[c*4+2], k3 = kw[c*4+3], kbc = kb[c];
      const float* s1 = S1 + b * 4; const float* s2 = S2 + b * 10;
      float d = k0*s1[0] + k1*s1[1] + k2*s1[2] + k3*s1[3];
      float q = k0*k0*s2[0] + 2.f*k0*k1*s2[1] + 2.f*k0*k2*s2[2] + 2.f*k0*k3*s2[3]
              + k1*k1*s2[4] + 2.f*k1*k2*s2[5] + 2.f*k1*k3*s2[6]
              + k2*k2*s2[7] + 2.f*k2*k3*s2[8] + k3*k3*s2[9];
      sy[c]  = d + kbc * CNT2;
      syy[c] = q + 2.f * kbc * d + kbc * kbc * CNT2;
    }
    __syncthreads();
    if (tid < 8) {
      float S = 0.f, Q = 0.f;
      for (int c = tid * 8; c < tid * 8 + 8; c++) { S += sy[c]; Q += syy[c]; }
      float mean = S / (8.f * CNT2);
      float var  = Q / (8.f * CNT2) - mean * mean;
      m2[tid] = mean; i2[tid] = rsqrtf(var + 1e-5f);
    }
    __syncthreads();
    {
      int c = tid, g = c >> 4;
      float a = i1[g] * gn1g[c];
      A1[b * 128 + c] = a;
      C1[b * 128 + c] = gn1b[c] - m1[g] * a;
    }
    if (tid < 64) {
      int c = tid, g = c >> 3;
      float a = i2[g] * gn2g[c];
      A2[b * 64 + c] = a;
      C2[b * 64 + c] = gn2b[c] - m2[g] * a;
    }
    __syncthreads();
  }
}

// ---------------- output kernel: KNN max + two WMMA GEMMs ----------------
__global__ __launch_bounds__(128) void kout(
    const float* __restrict__ x_pre, const float* __restrict__ fws,
    const _Float16* __restrict__ w2p, const _Float16* __restrict__ owp,
    const float* __restrict__ A1, const float* __restrict__ C1,
    const float* __restrict__ A2, const float* __restrict__ C2,
    const float* __restrict__ kw, const float* __restrict__ kb,
    const float* __restrict__ b2, const float* __restrict__ ob,
    const float* __restrict__ prelu1, const float* __restrict__ prelu2,
    float* __restrict__ out) {
  __shared__ float skw[256], skb[64], sA2[64], sC2[64], sA1[128], sC1[128], sbias[64];
  __shared__ float mbuf[4][16][64];
  int tid = threadIdx.x;
  int wave = tid >> 5, lane = tid & 31;
  int pblk = blockIdx.x * 64;            // 4 waves x 16 points
  int b = pblk >> 14;
  int nblk = pblk & (NN - 1);
  for (int t = tid; t < 256; t += 128) skw[t] = kw[t];
  if (tid < 64) {
    skb[tid] = kb[tid]; sA2[tid] = A2[b * 64 + tid]; sC2[tid] = C2[b * 64 + tid];
    sbias[tid] = b2[tid] + ob[tid];
  }
  sA1[tid] = A1[b * 128 + tid]; sC1[tid] = C1[b * 128 + tid];
  float p1 = prelu1[0], p2 = prelu2[0];
  __syncthreads();

  int pl = lane & 15, half = lane >> 4;
  size_t p = (size_t)pblk + wave * 16 + pl;

  // ---- stage 1: knn branch, 2 lanes per point, 32 channels each ----
  float mx[32];
  #pragma unroll
  for (int c = 0; c < 32; c++) mx[c] = -3.4e38f;
  const float4* fp = (const float4*)(fws + p * DKK * 4);
  for (int k = 0; k < DKK; k++) {
    float4 f = fp[k];
    #pragma unroll
    for (int c = 0; c < 32; c++) {
      int ch = half * 32 + c;
      float y = skw[ch*4]*f.x + skw[ch*4+1]*f.y + skw[ch*4+2]*f.z + skw[ch*4+3]*f.w + skb[ch];
      float z = sA2[ch] * y + sC2[ch];
      z = (z >= 0.f) ? z : p2 * z;
      mx[c] = fmaxf(mx[c], z);
    }
  }
  #pragma unroll
  for (int c = 0; c < 32; c++) mbuf[wave][pl][half * 32 + c] = mx[c];
  __syncthreads();

  // ---- stage 2: acc = h128 @ w2^T + m64 @ ow^T (WMMA f16->f32) ----
  v8f zero = {0.f,0.f,0.f,0.f,0.f,0.f,0.f,0.f};
  v8f acc[4];
  #pragma unroll
  for (int nt = 0; nt < 4; nt++) acc[nt] = zero;

  const float* xp = x_pre + p * 128;
  #pragma unroll
  for (int kt = 0; kt < 4; kt++) {
    v16h a;
    #pragma unroll
    for (int j = 0; j < 16; j++) {
      int c = kt * 32 + amap(half, j);
      float v = xp[c];
      v = sA1[c] * v + sC1[c];
      v = (v >= 0.f) ? v : p1 * v;
      a[j] = (_Float16)v;
    }
    #pragma unroll
    for (int nt = 0; nt < 4; nt++) {
      v16h bm = ((const v16h*)w2p)[(kt * 4 + nt) * 32 + lane];
      acc[nt] = __builtin_amdgcn_wmma_f32_16x16x32_f16(
          false, a, false, bm, (short)0, acc[nt], false, false);
    }
  }
  #pragma unroll
  for (int kt = 0; kt < 2; kt++) {
    v16h a;
    #pragma unroll
    for (int j = 0; j < 16; j++) {
      int c = kt * 32 + amap(half, j);
      a[j] = (_Float16)mbuf[wave][pl][c];
    }
    #pragma unroll
    for (int nt = 0; nt < 4; nt++) {
      v16h bm = ((const v16h*)owp)[(kt * 4 + nt) * 32 + lane];
      acc[nt] = __builtin_amdgcn_wmma_f32_16x16x32_f16(
          false, a, false, bm, (short)0, acc[nt], false, false);
    }
  }

  // ---- store: C layout lane L reg r -> point M=r+8*half, channel N=pl ----
  #pragma unroll
  for (int nt = 0; nt < 4; nt++) {
    int ch = nt * 16 + pl;
    float bias = sbias[ch];
    size_t orow = ((size_t)b * 64 + ch) * NN + nblk + wave * 16 + half * 8;
    float4 lo = make_float4(acc[nt][0]+bias, acc[nt][1]+bias, acc[nt][2]+bias, acc[nt][3]+bias);
    float4 hi = make_float4(acc[nt][4]+bias, acc[nt][5]+bias, acc[nt][6]+bias, acc[nt][7]+bias);
    *(float4*)(out + orow)     = lo;
    *(float4*)(out + orow + 4) = hi;
  }
}

extern "C" void kernel_launch(void* const* d_in, const int* in_sizes, int n_in,
                              void* d_out, int out_size, void* d_ws, size_t ws_size,
                              hipStream_t stream) {
  (void)in_sizes; (void)n_in; (void)out_size; (void)ws_size;
  const float* coords = (const float*)d_in[0];
  const float* tcorr  = (const float*)d_in[1];
  const float* xyz2   = (const float*)d_in[2];
  const float* dflow  = (const float*)d_in[3];
  const float* w1     = (const float*)d_in[4];
  const float* b1     = (const float*)d_in[5];
  const float* gn1g   = (const float*)d_in[6];
  const float* gn1b   = (const float*)d_in[7];
  const float* pre1   = (const float*)d_in[8];
  const float* w2     = (const float*)d_in[9];
  const float* b2     = (const float*)d_in[10];
  const float* kw     = (const float*)d_in[11];
  const float* kb     = (const float*)d_in[12];
  const float* gn2g   = (const float*)d_in[13];
  const float* gn2b   = (const float*)d_in[14];
  const float* pre2   = (const float*)d_in[15];
  const float* ow     = (const float*)d_in[16];
  const float* ob     = (const float*)d_in[17];
  float* out = (float*)d_out;

  char* ws = (char*)d_ws;
  size_t o = 0;
  auto carve = [&](size_t bytes) {
    char* r = ws + o;
    o = (o + bytes + 255) & ~(size_t)255;
    return r;
  };
  float*    x_pre = (float*)carve((size_t)NBP * 128 * 4);      // 16 MB
  float*    fws   = (float*)carve((size_t)NBP * DKK * 4 * 4);  // 14 MB
  float*    w1eff = (float*)carve(128 * 27 * 4);
  _Float16* w2p   = (_Float16*)carve(16 * 32 * 16 * 2);
  _Float16* owp   = (_Float16*)carve(8 * 32 * 16 * 2);
  float*    stats = (float*)carve(60 * 4);
  float* gsum = stats;       float* gsq = stats + 16;
  float* S1v  = stats + 32;  float* S2v = stats + 40;
  float* A1 = (float*)carve(256 * 4);
  float* C1 = (float*)carve(256 * 4);
  float* A2 = (float*)carve(128 * 4);
  float* C2 = (float*)carve(128 * 4);

  hipMemsetAsync(stats, 0, 60 * sizeof(float), stream);
  kprep<<<1, 256, 0, stream>>>(w1, w2, ow, w1eff, w2p, owp);
  kpoint<<<NBP / 8, 256, 0, stream>>>(coords, tcorr, xyz2, dflow, b1, w1eff,
                                      x_pre, fws, gsum, gsq, S1v, S2v);
  kfin<<<1, 128, 0, stream>>>(gsum, gsq, S1v, S2v, gn1g, gn1b, gn2g, gn2b,
                              kw, kb, A1, C1, A2, C2);
  kout<<<NBP / 64, 128, 0, stream>>>(x_pre, fws, w2p, owp, A1, C1, A2, C2,
                                     kw, kb, b2, ob, pre1, pre2, out);
}
